// GraphUpsampler_41652592837272
// MI455X (gfx1250) — matrix operand order repaired and checked by reading, not compiled, not run on hardware
//
#include <hip/hip_runtime.h>

typedef __attribute__((ext_vector_type(16))) __bf16 v16bf;
typedef __attribute__((ext_vector_type(8)))  __bf16 v8bf;
typedef __attribute__((ext_vector_type(8)))  float  v8f;

#define N_NODES 4096
#define M_NODES 8192
#define DIMF    128

// ---------------------------------------------------------------------------
// WMMA fragment loaders (v_wmma_f32_16x16x32_bf16 layouts, CDNA5 ISA 7.12.2)
// A (16x32, MxK): lane m=lane%16, half=lane/16; VGPR0-3 K=8*half+0..7,
//                 VGPR4-7 K=16+8*half+0..7  -> two 16B loads per lane.
// B (32x16, KxN): lane n=lane%16, half=lane/16; VGPR0-3 K=8*half+0..7,
//                 VGPR4-7 K=16+8*half+0..7 -> two 16B loads from BT[n][k].
// ---------------------------------------------------------------------------
__device__ __forceinline__ v16bf cat8(v8bf lo, v8bf hi) {
  return __builtin_shufflevector(lo, hi, 0, 1, 2, 3, 4, 5, 6, 7,
                                         8, 9, 10, 11, 12, 13, 14, 15);
}

__device__ __forceinline__ v16bf load_fragA(const __bf16* rowp, int half) {
  v8bf lo = *(const v8bf*)(rowp + 8 * half);
  v8bf hi = *(const v8bf*)(rowp + 16 + 8 * half);
  return cat8(lo, hi);
}

__device__ __forceinline__ v16bf load_fragB(const __bf16* rowp, int half) {
  v8bf lo = *(const v8bf*)(rowp + 8 * half);
  v8bf hi = *(const v8bf*)(rowp + 16 + 8 * half);
  return cat8(lo, hi);
}

__device__ __forceinline__ float sigmoidf_dev(float v) {
  return 1.0f / (1.0f + __expf(-v));
}

// Epilogue modes
enum { EPI_F32 = 0, EPI_SIG_BF16 = 1, EPI_SIG_F32 = 2, EPI_GCN = 3,
       EPI_ROWBIAS_BF16 = 4 };

// C[M,N] = A[M,K](bf16,row-major) @ B, with B given transposed: BT[N,K].
// Block = 256 threads = 8 waves (4 M-waves x 2 N-waves); wave tile 32x64;
// block tile 128x128. All of M, N, K are multiples of 32 (M,N of 128).
template <int EPI>
__global__ __launch_bounds__(256) void gemm_bf16_wmma(
    const __bf16* __restrict__ A, int lda,
    const __bf16* __restrict__ BT, int ldb, int K,
    float* __restrict__ Cf, __bf16* __restrict__ Cb, int ldc,
    const float* __restrict__ bias, const float* __restrict__ dinv) {
  const int lane = threadIdx.x & 31;
  const int wave = threadIdx.x >> 5;
  const int l16  = lane & 15;
  const int half = lane >> 4;
  const int row0 = blockIdx.x * 128 + (wave >> 1) * 32;
  const int col0 = blockIdx.y * 128 + (wave & 1) * 64;

  const __bf16* a0 = A + (size_t)(row0 + l16) * lda;
  const __bf16* a1 = A + (size_t)(row0 + 16 + l16) * lda;
  const __bf16* bp0 = BT + (size_t)(col0 + l16) * ldb;
  const __bf16* bp1 = BT + (size_t)(col0 + 16 + l16) * ldb;
  const __bf16* bp2 = BT + (size_t)(col0 + 32 + l16) * ldb;
  const __bf16* bp3 = BT + (size_t)(col0 + 48 + l16) * ldb;

  v8f acc[2][4];
  for (int i = 0; i < 2; ++i)
    for (int j = 0; j < 4; ++j)
      for (int r = 0; r < 8; ++r) acc[i][j][r] = 0.0f;

  for (int k = 0; k < K; k += 32) {
    if (k + 256 < K) {  // stream-ahead for the big-K (Adj) GEMMs
      // locality=3 -> WGP-scope prefetch: pulls into all cache levels,
      // which is right for an operand this wave itself consumes shortly.
      __builtin_prefetch(a0 + k + 256, 0, 3);
      __builtin_prefetch(a1 + k + 256, 0, 3);
    }
    v16bf af[2], bf[4];
    af[0] = load_fragA(a0 + k, half);
    af[1] = load_fragA(a1 + k, half);
    bf[0] = load_fragB(bp0 + k, half);
    bf[1] = load_fragB(bp1 + k, half);
    bf[2] = load_fragB(bp2 + k, half);
    bf[3] = load_fragB(bp3 + k, half);
#pragma unroll
    for (int i = 0; i < 2; ++i)
#pragma unroll
      for (int j = 0; j < 4; ++j)
        acc[i][j] = __builtin_amdgcn_wmma_f32_16x16x32_bf16(
            false, af[i], false, bf[j], (short)0, acc[i][j], false, false);
  }

#pragma unroll
  for (int i = 0; i < 2; ++i) {
#pragma unroll
    for (int j = 0; j < 4; ++j) {
#pragma unroll
      for (int r = 0; r < 8; ++r) {
        const int m = row0 + i * 16 + 8 * half + r;  // C/D: lanes16-31 -> M=8+r
        const int n = col0 + j * 16 + l16;
        float v = acc[i][j][r];
        if constexpr (EPI == EPI_F32) {
          Cf[(size_t)m * ldc + n] = v;
        } else if constexpr (EPI == EPI_SIG_BF16) {
          Cb[(size_t)m * ldc + n] = (__bf16)sigmoidf_dev(v);
        } else if constexpr (EPI == EPI_SIG_F32) {
          Cf[(size_t)m * ldc + n] = sigmoidf_dev(v);
        } else if constexpr (EPI == EPI_GCN) {
          v = fmaxf(v * dinv[m] + bias[n], 0.0f);  // relu(dinv_m*Z + b)
          Cb[(size_t)m * ldc + n] = (__bf16)v;
        } else {  // EPI_ROWBIAS_BF16 (upsample: + b_up[row])
          Cb[(size_t)m * ldc + n] = (__bf16)(v + bias[m]);
        }
      }
    }
  }
}

// ------------------------------- helpers -----------------------------------
__global__ void k_f32_to_bf16(const float* __restrict__ in,
                              __bf16* __restrict__ out, int n) {
  int i = blockIdx.x * blockDim.x + threadIdx.x;
  if (i < n) out[i] = (__bf16)in[i];
}

__global__ void k_transpose_f32_to_bf16(const float* __restrict__ in,
                                        __bf16* __restrict__ out, int rows,
                                        int cols) {
  int i = blockIdx.x * blockDim.x + threadIdx.x;
  if (i < rows * cols) {
    int r = i / cols, c = i % cols;
    out[(size_t)c * rows + r] = (__bf16)in[i];
  }
}

// Adj = [[A, A^T], [A, S]]; the S block is filled by a sigmoid-GEMM.
__global__ void k_build_adj(const float* __restrict__ A,
                            __bf16* __restrict__ Adj) {
  int j = blockIdx.y * blockDim.x + threadIdx.x;
  int i = blockIdx.x;
  float aij = A[(size_t)i * N_NODES + j];
  float aji = A[(size_t)j * N_NODES + i];
  Adj[(size_t)i * M_NODES + j] = (__bf16)aij;
  Adj[(size_t)i * M_NODES + N_NODES + j] = (__bf16)aji;
  Adj[(size_t)(N_NODES + i) * M_NODES + j] = (__bf16)aij;
}

// Adj is symmetric -> column degree == row sum. dinv = rsqrt(deg).
// Vectorized v8bf (16B) loads: 8192 cols / 8 = 1024 chunks / 256 threads.
__global__ void k_row_sum_rsqrt(const __bf16* __restrict__ Adj,
                                float* __restrict__ dinv, int ncols) {
  __shared__ float red[256];
  int row = blockIdx.x;
  const v8bf* p = (const v8bf*)(Adj + (size_t)row * ncols);
  const int nchunk = ncols >> 3;
  float s = 0.0f;
  for (int c = threadIdx.x; c < nchunk; c += 256) {
    v8bf v = p[c];
#pragma unroll
    for (int e = 0; e < 8; ++e) s += (float)v[e];
  }
  red[threadIdx.x] = s;
  __syncthreads();
  for (int offv = 128; offv > 0; offv >>= 1) {
    if (threadIdx.x < offv) red[threadIdx.x] += red[threadIdx.x + offv];
    __syncthreads();
  }
  if (threadIdx.x == 0) dinv[row] = rsqrtf(red[0]);
}

// YsT[n][k] = bf16(dinv[k] * XW[k][n])  (scale + transpose into BT layout)
__global__ void k_scale_transpose(const float* __restrict__ XW,
                                  const float* __restrict__ dinv,
                                  __bf16* __restrict__ YsT) {
  int i = blockIdx.x * blockDim.x + threadIdx.x;  // i < M_NODES*DIMF
  int k = i >> 7, n = i & 127;
  YsT[(size_t)n * M_NODES + k] = (__bf16)(XW[i] * dinv[k]);
}

// ------------------------------- driver ------------------------------------
extern "C" void kernel_launch(void* const* d_in, const int* in_sizes, int n_in,
                              void* d_out, int out_size, void* d_ws,
                              size_t ws_size, hipStream_t stream) {
  (void)in_sizes; (void)n_in; (void)out_size; (void)ws_size;
  const float* X    = (const float*)d_in[0];
  const float* Ain  = (const float*)d_in[1];
  const float* Wup  = (const float*)d_in[2];
  const float* b_up = (const float*)d_in[3];
  const float* W1   = (const float*)d_in[4];
  const float* b1   = (const float*)d_in[5];
  const float* W2   = (const float*)d_in[6];
  const float* b2   = (const float*)d_in[7];

  char* ws = (char*)d_ws;
  size_t off = 0;
  auto carve = [&](size_t bytes) {
    void* p = ws + off;
    off += (bytes + 255) & ~(size_t)255;
    return p;
  };
  __bf16* Adj  = (__bf16*)carve((size_t)M_NODES * M_NODES * 2);  // 128 MB
  __bf16* WupB = (__bf16*)carve((size_t)N_NODES * N_NODES * 2);  // 32 MB
  __bf16* XT   = (__bf16*)carve((size_t)DIMF * N_NODES * 2);
  __bf16* Xu   = (__bf16*)carve((size_t)M_NODES * DIMF * 2);
  float*  XW   = (float*)carve((size_t)M_NODES * DIMF * 4);
  __bf16* YsT  = (__bf16*)carve((size_t)DIMF * M_NODES * 2);
  __bf16* W1T  = (__bf16*)carve((size_t)DIMF * DIMF * 2);
  __bf16* W2T  = (__bf16*)carve((size_t)DIMF * DIMF * 2);
  float*  dinv = (float*)carve((size_t)M_NODES * 4);

  dim3 blk(256);
  k_f32_to_bf16<<<(N_NODES * N_NODES + 255) / 256, blk, 0, stream>>>(
      Wup, WupB, N_NODES * N_NODES);
  k_transpose_f32_to_bf16<<<(N_NODES * DIMF + 255) / 256, blk, 0, stream>>>(
      X, XT, N_NODES, DIMF);
  k_transpose_f32_to_bf16<<<(DIMF * DIMF + 255) / 256, blk, 0, stream>>>(
      W1, W1T, DIMF, DIMF);
  k_transpose_f32_to_bf16<<<(DIMF * DIMF + 255) / 256, blk, 0, stream>>>(
      W2, W2T, DIMF, DIMF);
  k_f32_to_bf16<<<(N_NODES * DIMF + 255) / 256, blk, 0, stream>>>(
      X, Xu, N_NODES * DIMF);

  // new_nodes = W_up @ X + b_up  -> Xu rows [N_NODES, M_NODES)
  gemm_bf16_wmma<EPI_ROWBIAS_BF16><<<dim3(N_NODES / 128, 1), blk, 0, stream>>>(
      WupB, N_NODES, XT, N_NODES, N_NODES, nullptr,
      Xu + (size_t)N_NODES * DIMF, DIMF, b_up, nullptr);

  // Adj = [[A, A^T], [A, sigmoid(new @ new^T)]]
  k_build_adj<<<dim3(N_NODES, N_NODES / 256), blk, 0, stream>>>(Ain, Adj);
  gemm_bf16_wmma<EPI_SIG_BF16>
      <<<dim3(N_NODES / 128, N_NODES / 128), blk, 0, stream>>>(
          Xu + (size_t)N_NODES * DIMF, DIMF, Xu + (size_t)N_NODES * DIMF, DIMF,
          DIMF, nullptr, Adj + (size_t)N_NODES * M_NODES + N_NODES, M_NODES,
          nullptr, nullptr);
  k_row_sum_rsqrt<<<M_NODES, blk, 0, stream>>>(Adj, dinv, M_NODES);

  for (int it = 0; it < 3; ++it) {
    // conv1: Xu = relu(dinv ⊙ (Adj @ (dinv ⊙ (Xu@W1))) + b1)
    gemm_bf16_wmma<EPI_F32><<<dim3(M_NODES / 128, 1), blk, 0, stream>>>(
        Xu, DIMF, W1T, DIMF, DIMF, XW, nullptr, DIMF, nullptr, nullptr);
    k_scale_transpose<<<(M_NODES * DIMF) / 256, blk, 0, stream>>>(XW, dinv,
                                                                  YsT);
    gemm_bf16_wmma<EPI_GCN><<<dim3(M_NODES / 128, 1), blk, 0, stream>>>(
        Adj, M_NODES, YsT, M_NODES, M_NODES, nullptr, Xu, DIMF, b1, dinv);
    // mid-loop rebuild: Adj = sigmoid(Xu @ Xu^T)
    gemm_bf16_wmma<EPI_SIG_BF16>
        <<<dim3(M_NODES / 128, M_NODES / 128), blk, 0, stream>>>(
            Xu, DIMF, Xu, DIMF, DIMF, nullptr, Adj, M_NODES, nullptr, nullptr);
    k_row_sum_rsqrt<<<M_NODES, blk, 0, stream>>>(Adj, dinv, M_NODES);
    // conv2 (Adj and dinv carry to the next iteration's conv1, per reference)
    gemm_bf16_wmma<EPI_F32><<<dim3(M_NODES / 128, 1), blk, 0, stream>>>(
        Xu, DIMF, W2T, DIMF, DIMF, XW, nullptr, DIMF, nullptr, nullptr);
    k_scale_transpose<<<(M_NODES * DIMF) / 256, blk, 0, stream>>>(XW, dinv,
                                                                  YsT);
    gemm_bf16_wmma<EPI_GCN><<<dim3(M_NODES / 128, 1), blk, 0, stream>>>(
        Adj, M_NODES, YsT, M_NODES, M_NODES, nullptr, Xu, DIMF, b2, dinv);
    // A_out of iterations 0 and 1 is dead (overwritten) -> only compute last.
  }
  gemm_bf16_wmma<EPI_SIG_F32>
      <<<dim3(M_NODES / 128, M_NODES / 128), blk, 0, stream>>>(
          Xu, DIMF, Xu, DIMF, DIMF, (float*)d_out, nullptr, M_NODES, nullptr,
          nullptr);
}
